// PersonalMed_68401649156269
// MI455X (gfx1250) — matrix operand order repaired and checked by reading, not compiled, not run on hardware
//
#include <hip/hip_runtime.h>
#include <math.h>

// Problem constants (match reference)
#define DD 128

typedef __bf16 bf16;
typedef bf16  v16bf __attribute__((ext_vector_type(16)));
typedef bf16  v8bf  __attribute__((ext_vector_type(8)));
typedef float v8f   __attribute__((ext_vector_type(8)));

__device__ __forceinline__ v8f wmma_bf(v16bf a, v16bf b, v8f c) {
  // D = A(16x32 bf16) * B(32x16 bf16) + C(16x16 f32)
  return __builtin_amdgcn_wmma_f32_16x16x32_bf16(false, a, false, b, (short)0, c,
                                                 false, false);
}
__device__ __forceinline__ float sigm(float x) { return 1.0f / (1.0f + expf(-x)); }

// ---------------------------------------------------------------------------
// WMMA GEMM:  C[M,N] = A[M,K] (bf16, row-major) * B (bf16, ALWAYS stored in
// the fragment-friendly [n][k] layout, ldb = K) (+bias +row-broadcast addmat,
// optional relu), store f32 and/or bf16.  KCH = K/32 is a template constant
// so the K-loop fully unrolls and loads get clause-batched ahead of the WMMAs.
// grid.x = M/16, grid.y = Ntiles/wavesPerBlock, blockDim = wavesPerBlock*32
// ---------------------------------------------------------------------------
template <int KCH>
__global__ __launch_bounds__(128) void gemm_wmma_kernel(
    const bf16* __restrict__ A, int lda,
    const bf16* __restrict__ Bm, int ldb,
    const float* __restrict__ bias,
    const float* __restrict__ addmat, int addld, int rshift,
    int do_relu,
    float* __restrict__ outF, int ldf,
    bf16* __restrict__ outB, int ldob,
    int Nvalid)
{
  const int lane  = threadIdx.x & 31;
  const int wave  = threadIdx.x >> 5;
  const int wpb   = blockDim.x >> 5;
  const int mtile = blockIdx.x;
  const int ntile = blockIdx.y * wpb + wave;   // wave-uniform
  const int h     = lane >> 4;
  const int l16   = lane & 15;

  v8f acc = {0.f,0.f,0.f,0.f,0.f,0.f,0.f,0.f};
  const bf16* arow = A  + (size_t)(mtile * 16 + l16) * lda + 8 * h;
  const bf16* brow = Bm + (size_t)(ntile * 16 + l16) * ldb + 8 * h;
#pragma unroll
  for (int kc = 0; kc < KCH; ++kc) {
    v8bf lo = *(const v8bf*)(arow + kc * 32);        // k = kbase+8h .. +7
    v8bf hi = *(const v8bf*)(arow + kc * 32 + 16);   // k = kbase+16+8h .. +7
    v16bf a = __builtin_shufflevector(lo, hi, 0,1,2,3,4,5,6,7,8,9,10,11,12,13,14,15);
    v8bf blo = *(const v8bf*)(brow + kc * 32);
    v8bf bhi = *(const v8bf*)(brow + kc * 32 + 16);
    v16bf b = __builtin_shufflevector(blo, bhi, 0,1,2,3,4,5,6,7,8,9,10,11,12,13,14,15);  // same k order as a
    acc = wmma_bf(a, b, acc);
  }
  const int col = ntile * 16 + l16;
  if (col < Nvalid) {
#pragma unroll
    for (int r = 0; r < 8; ++r) {
      const int row = mtile * 16 + 8 * h + r;
      float v = acc[r];
      if (bias)   v += bias[col];
      if (addmat) v += addmat[(size_t)(row >> rshift) * addld + col];
      if (do_relu) v = fmaxf(v, 0.0f);
      if (outF) outF[(size_t)row * ldf + col] = v;
      if (outB) outB[(size_t)row * ldob + col] = (bf16)v;
    }
  }
}

// ---------------------------------------------------------------------------
// Persistent GRU recurrence.  GI = x@Wih^T+bih precomputed ([seq*T+t, 384]).
// H lives in LDS (bf16, up to 256x128 = 64KB).  Each wave owns PPW
// (mtile,dtile) patches; per step it computes the r/z/g 16x16 gh tiles with
// WMMA (K=128), does fp32 gate math, and writes H back.
// mode 0: final h -> outB (bf16 [Nseq,128]);  mode 1: outAcc += final h.
// Whh is bf16 [384][128] (n-major -> contiguous-k B fragments).
// ---------------------------------------------------------------------------
template <int PPW>
__global__ __launch_bounds__(512) void gru_seq_kernel(
    const float* __restrict__ GI,
    const bf16* __restrict__ Whh,
    const float* __restrict__ bhh,
    int Nseq, int T, int mode,
    bf16* __restrict__ outB,
    float* __restrict__ outAcc)
{
  __shared__ bf16 Hs[256 * DD];
  const int tid  = threadIdx.x;
  const int lane = tid & 31;
  const int wave = tid >> 5;
  const int h    = lane >> 4;
  const int l16  = lane & 15;

  for (int i = tid; i < Nseq * DD; i += blockDim.x) Hs[i] = (bf16)0.0f;
  __syncthreads();

  float hnew[PPW][8];
  for (int t = 0; t < T; ++t) {
#pragma unroll
    for (int j = 0; j < PPW; ++j) {
      const int p = wave * PPW + j;
      const int mtile = p >> 3, dtile = p & 7;
      const int d = dtile * 16 + l16;
      v8f aR = {0.f,0.f,0.f,0.f,0.f,0.f,0.f,0.f};
      v8f aZ = aR, aG = aR;
      const bf16* hrow = &Hs[(mtile * 16 + l16) * DD + 8 * h];
#pragma unroll
      for (int kc = 0; kc < 4; ++kc) {
        v8bf lo = *(const v8bf*)(hrow + kc * 32);
        v8bf hi = *(const v8bf*)(hrow + kc * 32 + 16);
        v16bf a = __builtin_shufflevector(lo, hi, 0,1,2,3,4,5,6,7,8,9,10,11,12,13,14,15);
        const bf16* wb = Whh + kc * 32 + 8 * h;
        const bf16* w0 = wb + (size_t)(d) * DD;
        const bf16* w1 = wb + (size_t)(DD + d) * DD;
        const bf16* w2 = wb + (size_t)(2 * DD + d) * DD;
        v16bf b0 = __builtin_shufflevector(*(const v8bf*)w0, *(const v8bf*)(w0 + 16), 0,1,2,3,4,5,6,7,8,9,10,11,12,13,14,15);
        v16bf b1 = __builtin_shufflevector(*(const v8bf*)w1, *(const v8bf*)(w1 + 16), 0,1,2,3,4,5,6,7,8,9,10,11,12,13,14,15);
        v16bf b2 = __builtin_shufflevector(*(const v8bf*)w2, *(const v8bf*)(w2 + 16), 0,1,2,3,4,5,6,7,8,9,10,11,12,13,14,15);
        aR = wmma_bf(a, b0, aR);
        aZ = wmma_bf(a, b1, aZ);
        aG = wmma_bf(a, b2, aG);
      }
      const float br = bhh[d], bz = bhh[DD + d], bg = bhh[2 * DD + d];
#pragma unroll
      for (int r = 0; r < 8; ++r) {
        const int seq = mtile * 16 + 8 * h + r;
        const size_t gix = ((size_t)seq * T + t) * 384;
        const float rr = sigm(GI[gix + d] + aR[r] + br);
        const float zz = sigm(GI[gix + DD + d] + aZ[r] + bz);
        const float nn = tanhf(GI[gix + 2 * DD + d] + rr * (aG[r] + bg));
        const float ho = (float)Hs[seq * DD + d];
        hnew[j][r] = (1.0f - zz) * nn + zz * ho;
      }
    }
    __syncthreads();
    if (t == T - 1) {
#pragma unroll
      for (int j = 0; j < PPW; ++j) {
        const int p = wave * PPW + j;
        const int d = (p & 7) * 16 + l16;
#pragma unroll
        for (int r = 0; r < 8; ++r) {
          const int seq = (p >> 3) * 16 + 8 * h + r;
          if (mode == 0) outB[(size_t)seq * DD + d] = (bf16)hnew[j][r];
          else           outAcc[(size_t)seq * DD + d] += hnew[j][r];
        }
      }
    } else {
#pragma unroll
      for (int j = 0; j < PPW; ++j) {
        const int p = wave * PPW + j;
        const int d = (p & 7) * 16 + l16;
#pragma unroll
        for (int r = 0; r < 8; ++r) {
          const int seq = (p >> 3) * 16 + 8 * h + r;
          Hs[seq * DD + d] = (bf16)hnew[j][r];
        }
      }
      __syncthreads();
    }
  }
}

// ------------------------- elementwise helpers -----------------------------
__global__ void cvt_bf16_kernel(const float* __restrict__ src, bf16* __restrict__ dst,
                                int nsrc, int ndst, float scale)
{
  int i = blockIdx.x * 256 + threadIdx.x;
  if (i < ndst) dst[i] = (bf16)(i < nsrc ? src[i] * scale : 0.0f);
}

// Batched 128x128 transpose-convert: src is nmat matrices [k][n] f32;
// dst gets [n][k] bf16 so GEMM B fragments are contiguous-k loads.
__global__ void cvt_bf16_T128_kernel(const float* __restrict__ src,
                                     bf16* __restrict__ dst, int nmat, float scale)
{
  int i = blockIdx.x * 256 + threadIdx.x;
  if (i < nmat * 16384) {
    const int mat = i >> 14, rem = i & 16383;
    const int k = rem >> 7, n = rem & 127;
    dst[(size_t)(mat << 14) + n * 128 + k] = (bf16)(src[i] * scale);
  }
}

__global__ void zero_kernel(float* p, int n)
{
  int i = blockIdx.x * 256 + threadIdx.x;
  if (i < n) p[i] = 0.0f;
}

// vis[bv][d] = sum_c emb[codes[bv,c]][d]   (grid = B*V blocks, 128 threads)
__global__ void embed_sum_kernel(const int* __restrict__ codes,
                                 const float* __restrict__ emb,
                                 float* __restrict__ out, int ncodes)
{
  const int bv = blockIdx.x, d = threadIdx.x;
  float s = 0.0f;
  for (int c = 0; c < ncodes; ++c)
    s += emb[(size_t)codes[bv * ncodes + c] * DD + d];
  out[bv * DD + d] = s;
}

// mon[row][d] = sum_c ei[ci]*ev[cv]  (row = (b*V+v)*M + m; grid = 8192)
__global__ void mon_kernel(const int* __restrict__ ci, const int* __restrict__ cv,
                           const float* __restrict__ ei, const float* __restrict__ ev,
                           bf16* __restrict__ out)
{
  const int row = blockIdx.x, d = threadIdx.x;
  float s = 0.0f;
#pragma unroll
  for (int c = 0; c < 16; ++c) {
    const int a = ci[row * 16 + c];
    const int b = cv[row * 16 + c];
    s += ei[(size_t)a * DD + d] * ev[(size_t)b * DD + d];
  }
  out[(size_t)row * DD + d] = (bf16)s;
}

// From vis[3][256*128] build bf16 A-matrices: avg = sum/3, self_t = vis_t,
// msg_t = (sum - vis_t)/3.
__global__ void gnn_prep_kernel(const float* __restrict__ vis,
                                bf16* __restrict__ avgA,
                                bf16* __restrict__ selfA,
                                bf16* __restrict__ msgA)
{
  const int i = blockIdx.x * 128 + threadIdx.x;  // < 32768
  const float v0 = vis[i], v1 = vis[32768 + i], v2 = vis[65536 + i];
  const float s = v0 + v1 + v2;
  const float inv3 = 1.0f / 3.0f;
  avgA[i] = (bf16)(s * inv3);
  selfA[i] = (bf16)v0; selfA[32768 + i] = (bf16)v1; selfA[65536 + i] = (bf16)v2;
  msgA[i] = (bf16)((s - v0) * inv3);
  msgA[32768 + i] = (bf16)((s - v1) * inv3);
  msgA[65536 + i] = (bf16)((s - v2) * inv3);
}

// xe[bv][d] = (x[bv]*W[d] + b[d]) * (x != 0)
__global__ void scalar_feat_kernel(const float* __restrict__ x,
                                   const float* __restrict__ W,
                                   const float* __restrict__ b,
                                   bf16* __restrict__ out)
{
  const int bv = blockIdx.x, d = threadIdx.x;
  const float xv = x[bv];
  const float m = (xv != 0.0f) ? 1.0f : 0.0f;
  out[bv * DD + d] = (bf16)((xv * W[d] + b[d]) * m);
}

// pe[b][key*128+d] = relu(hid[key][b][d])  -> bf16 [16,896]
__global__ void pe_relu_kernel(const float* __restrict__ hid, bf16* __restrict__ pe)
{
  const int i = blockIdx.x * 128 + threadIdx.x;  // < 16*896
  const int b = i / 896, k = i % 896;
  const int key = k >> 7, d = k & 127;
  pe[i] = (bf16)fmaxf(hid[(size_t)(key * 16 + b) * DD + d], 0.0f);
}

// ---------------------------------------------------------------------------
extern "C" void kernel_launch(void* const* d_in, const int* in_sizes, int n_in,
                              void* d_out, int out_size, void* d_ws, size_t ws_size,
                              hipStream_t stream)
{
  (void)in_sizes; (void)n_in; (void)out_size;
  const int*   codes_cond = (const int*)d_in[0];
  const int*   codes_proc = (const int*)d_in[1];
  const int*   codes_drug = (const int*)d_in[2];
  const float* weight_x = (const float*)d_in[7];
  const float* age_x    = (const float*)d_in[8];
  const float* emb_cond = (const float*)d_in[9];
  const float* emb_proc = (const float*)d_in[10];
  const float* emb_drug = (const float*)d_in[11];
  const float* mWih_f = (const float*)d_in[16];
  const float* mWhh_f = (const float*)d_in[17];
  const float* mbih   = (const float*)d_in[18];
  const float* mbhh   = (const float*)d_in[19];
  const float* vWih_f = (const float*)d_in[20];
  const float* vWhh_f = (const float*)d_in[21];
  const float* vbih   = (const float*)d_in[22];
  const float* vbhh   = (const float*)d_in[23];
  const float* gWself = (const float*)d_in[24];
  const float* gWmsg  = (const float*)d_in[25];
  const float* fcwW   = (const float*)d_in[26];
  const float* fcwB   = (const float*)d_in[27];
  const float* fcaW   = (const float*)d_in[28];
  const float* fcaB   = (const float*)d_in[29];
  const float* headW_f = (const float*)d_in[30];
  const float* headB   = (const float*)d_in[31];

  char* base = (char*)d_ws;
  size_t off = 0;
  auto alloc = [&](size_t bytes) -> void* {
    void* p = base + off;
    off += (bytes + 255) & ~(size_t)255;
    return p;
  };
  float* vis   = (float*)alloc((size_t)3 * 32768 * 4);
  bf16*  avgA  = (bf16*) alloc((size_t)32768 * 2);
  bf16*  selfA = (bf16*) alloc((size_t)3 * 32768 * 2);
  bf16*  msgA  = (bf16*) alloc((size_t)3 * 32768 * 2);
  bf16*  wself = (bf16*) alloc((size_t)8 * 16384 * 2);  // transposed [n][k]
  bf16*  wmsg  = (bf16*) alloc((size_t)8 * 16384 * 2);  // transposed [n][k]
  bf16*  wmsg3 = (bf16*) alloc((size_t)8 * 16384 * 2);  // transposed, /3
  bf16*  mWih  = (bf16*) alloc((size_t)5 * 49152 * 2);  // already [n][k]
  bf16*  mWhh  = (bf16*) alloc((size_t)5 * 49152 * 2);
  bf16*  vWih  = (bf16*) alloc((size_t)4 * 49152 * 2);
  bf16*  vWhh  = (bf16*) alloc((size_t)4 * 49152 * 2);
  bf16*  headW = (bf16*) alloc((size_t)608 * 896 * 2);  // already [n][k]
  bf16*  monb  = (bf16*) alloc((size_t)8192 * 128 * 2);
  float* Rm    = (float*)alloc((size_t)4 * 32768 * 4);
  bf16*  outsb = (bf16*) alloc((size_t)4 * 8192 * 128 * 2);
  float* GI    = (float*)alloc((size_t)8192 * 384 * 4);
  bf16*  mlast = (bf16*) alloc((size_t)32768 * 2);
  float* GIv   = (float*)alloc((size_t)256 * 384 * 4);
  bf16*  xeb   = (bf16*) alloc((size_t)32768 * 2);
  float* hid   = (float*)alloc((size_t)7 * 2048 * 4);
  bf16*  peb   = (bf16*) alloc((size_t)16 * 896 * 2);
  if (off > ws_size) return;  // insufficient workspace

  auto cvt = [&](const float* s, bf16* dst, int nsrc, int ndst, float scale) {
    cvt_bf16_kernel<<<(ndst + 255) / 256, 256, 0, stream>>>(s, dst, nsrc, ndst, scale);
  };
  // GNN weights: transpose [k][n] -> [n][k] while converting (X@W form)
  cvt_bf16_T128_kernel<<<(8 * 16384 + 255) / 256, 256, 0, stream>>>(gWself, wself, 8, 1.0f);
  cvt_bf16_T128_kernel<<<(8 * 16384 + 255) / 256, 256, 0, stream>>>(gWmsg,  wmsg,  8, 1.0f);
  cvt_bf16_T128_kernel<<<(8 * 16384 + 255) / 256, 256, 0, stream>>>(gWmsg,  wmsg3, 8, 1.0f / 3.0f);
  // GRU / head weights are [N,K] already (X@W^T form): plain convert
  cvt(mWih_f, mWih, 5 * 49152, 5 * 49152, 1.0f);
  cvt(mWhh_f, mWhh, 5 * 49152, 5 * 49152, 1.0f);
  cvt(vWih_f, vWih, 4 * 49152, 4 * 49152, 1.0f);
  cvt(vWhh_f, vWhh, 4 * 49152, 4 * 49152, 1.0f);
  cvt(headW_f, headW, 600 * 896, 608 * 896, 1.0f);  // zero-padded to 608 rows
  zero_kernel<<<(7 * 2048 + 255) / 256, 256, 0, stream>>>(hid, 7 * 2048);

  embed_sum_kernel<<<256, 128, 0, stream>>>(codes_cond, emb_cond, vis, 64);
  embed_sum_kernel<<<256, 128, 0, stream>>>(codes_proc, emb_proc, vis + 32768, 64);
  embed_sum_kernel<<<256, 128, 0, stream>>>(codes_drug, emb_drug, vis + 65536, 64);
  gnn_prep_kernel<<<256, 128, 0, stream>>>(vis, avgA, selfA, msgA);

  auto gemm = [&](const bf16* A, int lda, const bf16* Bm, int ldb,
                  const float* bias, const float* addmat, int addld, int rshift,
                  int relu, float* oF, int ldf, bf16* oB, int ldob,
                  int Mtiles, int Ntiles, int Nvalid, int Kchunks, int wpb) {
    dim3 g(Mtiles, Ntiles / wpb);
    if (Kchunks == 4)
      gemm_wmma_kernel<4><<<g, wpb * 32, 0, stream>>>(A, lda, Bm, ldb, bias,
          addmat, addld, rshift, relu, oF, ldf, oB, ldob, Nvalid);
    else
      gemm_wmma_kernel<28><<<g, wpb * 32, 0, stream>>>(A, lda, Bm, ldb, bias,
          addmat, addld, rshift, relu, oF, ldf, oB, ldob, Nvalid);
  };

  const int gidx_tab[2][4] = {{0, 2, 3, 4}, {1, 2, 3, 4}};
  const int hkey_tab[2][4] = {{0, 1, 2, 3}, {4, 1, 2, 3}};
  const int*   ci_tab[2] = {(const int*)d_in[3], (const int*)d_in[5]};
  const int*   cv_tab[2] = {(const int*)d_in[4], (const int*)d_in[6]};
  const float* ei_tab[2] = {(const float*)d_in[12], (const float*)d_in[14]};
  const float* ev_tab[2] = {(const float*)d_in[13], (const float*)d_in[15]};

  for (int p = 0; p < 2; ++p) {
    mon_kernel<<<8192, 128, 0, stream>>>(ci_tab[p], cv_tab[p], ei_tab[p], ev_tab[p], monb);

    // Row-broadcast terms R_t  (256-row GEMMs)
    gemm(avgA, 128, wmsg + p * 4 * 16384, 128, nullptr, nullptr, 0, 0, 0,
         Rm, 128, nullptr, 0, 16, 8, 128, 4, 4);
    for (int t = 1; t < 4; ++t) {
      gemm(selfA + (t - 1) * 32768, 128, wself + (p * 4 + t) * 16384, 128,
           nullptr, nullptr, 0, 0, 0, Rm + t * 32768, 128, nullptr, 0, 16, 8, 128, 4, 4);
      gemm(msgA + (t - 1) * 32768, 128, wmsg + (p * 4 + t) * 16384, 128,
           nullptr, Rm + t * 32768, 128, 0, 0, Rm + t * 32768, 128, nullptr, 0, 16, 8, 128, 4, 4);
    }

    for (int t = 0; t < 4; ++t) {
      const bf16* Wbig = (t == 0) ? (wself + p * 4 * 16384) : (wmsg3 + (p * 4 + t) * 16384);
      bf16* outs_t = outsb + (size_t)t * 8192 * 128;
      // out_t = relu( mon @ Wbig + R_t[row>>5] )   [8192,128] -> bf16
      gemm(monb, 128, Wbig, 128, nullptr, Rm + t * 32768, 128, 5, 1,
           nullptr, 0, outs_t, 128, 512, 8, 128, 4, 4);
      const int g = gidx_tab[p][t];
      // GI = out_t @ Wih^T + bih   [8192,384] f32
      gemm(outs_t, 128, mWih + g * 49152, 128, mbih + g * 384, nullptr, 0, 0, 0,
           GI, 384, nullptr, 0, 512, 24, 384, 4, 4);
      // Monitor-level recurrence (256 seqs, T=32) -> mlast
      gru_seq_kernel<8><<<1, 512, 0, stream>>>(GI, mWhh + g * 49152, mbhh + g * 384,
                                               256, 32, 0, mlast, nullptr);
      // Visit-level GI and recurrence (16 seqs, T=16) -> hid[key] +=
      gemm(mlast, 128, vWih + p * 49152, 128, vbih + p * 384, nullptr, 0, 0, 0,
           GIv, 384, nullptr, 0, 16, 24, 384, 4, 4);
      gru_seq_kernel<2><<<1, 128, 0, stream>>>(GIv, vWhh + p * 49152, vbhh + p * 384,
                                               16, 16, 1, nullptr,
                                               hid + hkey_tab[p][t] * 2048);
    }
  }

  // weight / age scalar features
  scalar_feat_kernel<<<256, 128, 0, stream>>>(weight_x, fcwW, fcwB, xeb);
  gemm(xeb, 128, vWih + 2 * 49152, 128, vbih + 2 * 384, nullptr, 0, 0, 0,
       GIv, 384, nullptr, 0, 16, 24, 384, 4, 4);
  gru_seq_kernel<2><<<1, 128, 0, stream>>>(GIv, vWhh + 2 * 49152, vbhh + 2 * 384,
                                           16, 16, 1, nullptr, hid + 5 * 2048);
  scalar_feat_kernel<<<256, 128, 0, stream>>>(age_x, fcaW, fcaB, xeb);
  gemm(xeb, 128, vWih + 3 * 49152, 128, vbih + 3 * 384, nullptr, 0, 0, 0,
       GIv, 384, nullptr, 0, 16, 24, 384, 4, 4);
  gru_seq_kernel<2><<<1, 128, 0, stream>>>(GIv, vWhh + 3 * 49152, vbhh + 3 * 384,
                                           16, 16, 1, nullptr, hid + 6 * 2048);

  // Concatenate + ReLU + output head  [16,896] @ [896,600]^T + b
  pe_relu_kernel<<<112, 128, 0, stream>>>(hid, peb);
  gemm(peb, 896, headW, 896, headB, nullptr, 0, 0, 0,
       (float*)d_out, 600, nullptr, 0, 1, 38, 600, 28, 2);
}
